// SparseMultiHeadAttention_74363063763372
// MI455X (gfx1250) — matrix operand order repaired and checked by reading, not compiled, not run on hardware
//
#include <hip/hip_runtime.h>
#include <hip/hip_bf16.h>

typedef __bf16 bf16;
typedef __attribute__((ext_vector_type(16))) __bf16 v16bf;
typedef __attribute__((ext_vector_type(8)))  __bf16 v8bf;
typedef __attribute__((ext_vector_type(8)))  float  v8f;
typedef __attribute__((ext_vector_type(4)))  int    v4i;

#define NHEAD   16
#define NACT    4
#define DMODEL  1024
#define DATTN   64
#define BATCH   2
#define SEQ     4096

#if defined(__has_builtin)
#  if __has_builtin(__builtin_amdgcn_global_load_async_to_lds_b128)
#    define HAVE_ASYNC_LDS 1
#  endif
#endif
#ifndef HAVE_ASYNC_LDS
#  define HAVE_ASYNC_LDS 0
#endif

// ---------------------------------------------------------------- WMMA core
static __device__ __forceinline__ v8f wmma_bf16(v16bf a, v16bf b, v8f c) {
  // D = A(16x32 bf16) x B(32x16 bf16) + C(16x16 f32)
  return __builtin_amdgcn_wmma_f32_16x16x32_bf16(
      false, a, false, b, (short)0, c, false, false);
}

// A-tile (16x32) from row-major bf16 (global or LDS).
// lane = row (lane&15); element e -> K = k0 + 16*half + e
static __device__ __forceinline__ v16bf load_a_bf16(const bf16* src,
                                                    int ldm, int row0, int k0, int lane) {
  int m = row0 + (lane & 15), half = lane >> 4;
  const bf16* p = src + (size_t)m * ldm + k0 + 16 * half;
  v8bf lo = *(const v8bf*)p;
  v8bf hi = *(const v8bf*)(p + 8);
  v16bf a;
#pragma unroll
  for (int e = 0; e < 8; ++e) { a[e] = lo[e]; a[8 + e] = hi[e]; }
  return a;
}

// B-tile (32x16) when the source is K-contiguous per column (layout [col][k]):
// lane = col (lane&15); element e -> K = k0 + 16*half + e  => one 32B load.
static __device__ __forceinline__ v16bf load_b_kcontig(const bf16* src,
                                                       size_t ldk, int k0, int n0, int lane) {
  int col = n0 + (lane & 15), half = lane >> 4;
  return *(const v16bf*)(src + (size_t)col * ldk + k0 + 16 * half);
}

// ---------------------------------------------------------------- async staging helpers
static __device__ __forceinline__ void copy16(const bf16* g, bf16* l) {
#if HAVE_ASYNC_LDS
  __builtin_amdgcn_global_load_async_to_lds_b128(
      (__attribute__((address_space(1))) v4i*)g,
      (__attribute__((address_space(3))) v4i*)l, 0, 0);
#else
  *(v8bf*)l = *(const v8bf*)g;
#endif
}

static __device__ __forceinline__ void wait_async() {
#if HAVE_ASYNC_LDS
#  if defined(__has_builtin) && __has_builtin(__builtin_amdgcn_s_wait_asynccnt)
  __builtin_amdgcn_s_wait_asynccnt(0);
#  else
  asm volatile("s_wait_asynccnt 0" ::: "memory");
#  endif
#endif
}

// ---------------------------------------------------------------- prep: x -> bf16
__global__ void k_xcvt(const float* __restrict__ x, bf16* __restrict__ xb) {
  size_t i = ((size_t)blockIdx.x * blockDim.x + threadIdx.x) * 8;
  const float4* p = (const float4*)(x + i);
  float4 f0 = p[0], f1 = p[1];
  v8bf o;
  o[0] = (bf16)f0.x; o[1] = (bf16)f0.y; o[2] = (bf16)f0.z; o[3] = (bf16)f0.w;
  o[4] = (bf16)f1.x; o[5] = (bf16)f1.y; o[6] = (bf16)f1.z; o[7] = (bf16)f1.w;
  *(v8bf*)(xb + i) = o;
}

// ---------------------------------------------------------------- kernel: mean over S
__global__ void k_meanx(const float* __restrict__ x, float* __restrict__ meanx) {
  int idx = blockIdx.x * blockDim.x + threadIdx.x;   // 0 .. B*D-1
  int b = idx / DMODEL, d = idx % DMODEL;
  const float* p = x + (size_t)b * SEQ * DMODEL + d;
  float s = 0.f;
  for (int t = 0; t < SEQ; ++t) s += p[(size_t)t * DMODEL];
  meanx[idx] = s * (1.0f / SEQ);
}

// ---------------------------------------------------------------- kernel: router (tiny)
__global__ void k_router(const float* __restrict__ meanx,
                         const float* __restrict__ Wr, const float* __restrict__ br,
                         int* __restrict__ top_idx, float* __restrict__ gate) {
  __shared__ float r[BATCH][NHEAD];
  int t = threadIdx.x;
  if (t < BATCH * NHEAD) {
    int b = t / NHEAD, h = t % NHEAD;
    float s = br[h];
    for (int d = 0; d < DMODEL; ++d) s += meanx[b * DMODEL + d] * Wr[d * NHEAD + h];
    r[b][h] = s;
  }
  __syncthreads();
  if (t < BATCH) {
    int b = t;
    float mx = r[b][0];
    for (int h = 1; h < NHEAD; ++h) mx = fmaxf(mx, r[b][h]);
    float dist[NHEAD], den = 0.f;
    for (int h = 0; h < NHEAD; ++h) { dist[h] = __expf(r[b][h] - mx); den += dist[h]; }
    for (int h = 0; h < NHEAD; ++h) dist[h] /= den;
    int   idxs[NACT];
    float vals[NACT];
    bool  used[NHEAD];
    for (int h = 0; h < NHEAD; ++h) used[h] = false;
    for (int k = 0; k < NACT; ++k) {
      int bi = -1; float bv = -1.f;
      for (int h = 0; h < NHEAD; ++h)
        if (!used[h] && dist[h] > bv) { bv = dist[h]; bi = h; }
      used[bi] = true; idxs[k] = bi; vals[k] = bv;
    }
    float gden = (float)(NHEAD - NACT);   // exp(0) for the 12 zeroed heads
    float ge[NACT];
    for (int k = 0; k < NACT; ++k) { ge[k] = __expf(vals[k]); gden += ge[k]; }
    for (int k = 0; k < NACT; ++k) {
      top_idx[b * NACT + k] = idxs[k];
      gate[b * NACT + k]    = ge[k] / gden;
    }
  }
}

// ---------------------------------------------------------------- prep: gather active W columns, transposed
// Wg[b][p][c=slot*64+d][kdim] = Wsel[kdim][top_idx[b,slot]*64 + d]  (bf16)
__global__ void k_wgather(const float* __restrict__ Wq, const float* __restrict__ Wk,
                          const float* __restrict__ Wv, const int* __restrict__ top_idx,
                          bf16* __restrict__ Wg) {
  int t = blockIdx.x * blockDim.x + threadIdx.x;  // 2*3*256*256 threads, 4 k each
  int kq = (t & 255) * 4;
  int c  = (t >> 8) & 255;
  int pb = t >> 16;                 // 0..5
  int p = pb % 3, b = pb / 3;
  const float* W = (p == 0) ? Wq : ((p == 1) ? Wk : Wv);
  int col = top_idx[b * NACT + (c >> 6)] * DATTN + (c & 63);
  bf16* dst = Wg + (((size_t)(b * 3 + p) * 256 + c) * DMODEL + kq);
#pragma unroll
  for (int j = 0; j < 4; ++j)
    dst[j] = (bf16)W[(size_t)(kq + j) * (NHEAD * DATTN) + col];
}

// ---------------------------------------------------------------- prep: transpose Wo -> [n][k] bf16
__global__ void k_wot(const float* __restrict__ Wo, bf16* __restrict__ Wot) {
  int t = blockIdx.x * blockDim.x + threadIdx.x;  // 1024*64 threads, 4 k each
  int n = t >> 6, kq = (t & 63) * 4;
#pragma unroll
  for (int j = 0; j < 4; ++j)
    Wot[(size_t)n * (NACT * DATTN) + kq + j] = (bf16)Wo[(size_t)(kq + j) * DMODEL + n];
}

// ---------------------------------------------------------------- gated QKV (active heads only)
// one wave per (b,slot,proj, 16-row tile); computes full 16x64 head tile (4 accumulators / shared A)
__global__ __launch_bounds__(128) void k_qkv(
    const bf16* __restrict__ xb, const bf16* __restrict__ Wg,
    const float* __restrict__ bq, const float* __restrict__ bk, const float* __restrict__ bv,
    const int* __restrict__ top_idx, const float* __restrict__ gate,
    bf16* __restrict__ Qs, bf16* __restrict__ Ks, bf16* __restrict__ Vt) {
  int wave = (blockIdx.x * blockDim.x + threadIdx.x) >> 5;
  int lane = threadIdx.x & 31;
  int mt = wave & 255;               // 256 tiles of 16 across SEQ
  int c  = wave >> 8;                // 0..23 = (b, slot, proj)
  int b = c / 12, k = (c / 3) % 4, p = c % 3;

  const bf16* xrow = xb + (size_t)b * SEQ * DMODEL;
  const bf16* Wcp  = Wg + ((size_t)(b * 3 + p) * 256 + k * 64) * DMODEL;

  v8f acc[4] = {};
  for (int kk = 0; kk < DMODEL; kk += 32) {
    v16bf a = load_a_bf16(xrow, DMODEL, mt * 16, kk, lane);
#pragma unroll
    for (int j = 0; j < 4; ++j) {
      v16bf bm = load_b_kcontig(Wcp, DMODEL, kk, j * 16, lane);
      acc[j] = wmma_bf16(a, bm, acc[j]);
    }
  }

  int   half = lane >> 4, n = lane & 15;
  int   h = top_idx[b * NACT + k];
  float g = gate[b * NACT + k];
  const float* bias = (p == 0) ? bq : ((p == 1) ? bk : bv);

  if (p < 2) {  // Q, K: row-major [s][64]
    bf16* dp = ((p == 0) ? Qs : Ks) + (size_t)(b * NACT + k) * SEQ * DATTN;
#pragma unroll
    for (int j = 0; j < 4; ++j) {
      float bb = bias[h * DATTN + j * 16 + n];
#pragma unroll
      for (int r = 0; r < 8; ++r) {
        int row = mt * 16 + 8 * half + r;
        dp[(size_t)row * DATTN + j * 16 + n] = (bf16)((acc[j][r] + bb) * g);
      }
    }
  } else {      // V: transposed [d][s] so attention's B-tile is one 32B load
    bf16* dp = Vt + (size_t)(b * NACT + k) * DATTN * SEQ;
#pragma unroll
    for (int j = 0; j < 4; ++j) {
      float bb = bias[h * DATTN + j * 16 + n];
#pragma unroll
      for (int r = 0; r < 8; ++r) {
        int row = mt * 16 + 8 * half + r;
        dp[(size_t)(j * 16 + n) * SEQ + row] = (bf16)((acc[j][r] + bb) * g);
      }
    }
  }
}

// ---------------------------------------------------------------- flash attention
// block = 4 waves = 4 query tiles of one (b,slot); K/V tiles double-buffered in LDS,
// staged with async global->LDS copies (ASYNCcnt) when available.
__global__ __launch_bounds__(128) void k_attn(
    const bf16* __restrict__ Qs, const bf16* __restrict__ Ks,
    const bf16* __restrict__ Vt, bf16* __restrict__ AO) {
  __shared__ __align__(32) bf16 Kt_s[2][32 * 64];   // [buf][t_rel][d]
  __shared__ __align__(32) bf16 Vt_s[2][64 * 32];   // [buf][d][t_rel]
  __shared__ __align__(32) bf16 pshare[4][16 * 32]; // per-wave P tile

  int tid   = threadIdx.x;
  int wslot = tid >> 5;
  int lane  = tid & 31;
  int c  = blockIdx.x >> 6;                 // 0..7 = b*4 + slot
  int mt = (blockIdx.x & 63) * 4 + wslot;   // query tile 0..255
  int b = c >> 2, k = c & 3;

  const bf16* Q  = Qs + (size_t)c * SEQ * DATTN;
  const bf16* K  = Ks + (size_t)c * SEQ * DATTN;
  const bf16* Vg = Vt + (size_t)c * DATTN * SEQ;
  int half = lane >> 4, n = lane & 15;

  v16bf qa0 = load_a_bf16(Q, DATTN, mt * 16, 0, lane);
  v16bf qa1 = load_a_bf16(Q, DATTN, mt * 16, 32, lane);

  float m[8], l[8];
  v8f o0 = {}, o1 = {}, o2 = {}, o3 = {};
#pragma unroll
  for (int r = 0; r < 8; ++r) { m[r] = -1e30f; l[r] = 0.f; }
  bf16* P = pshare[wslot];
  const float sc = 0.125f;  // 1/sqrt(64)

  // staging indices: 16B granules; 128 threads x 2 granules cover each 4KB tile
  int kr0 = tid >> 3,          ks0 = tid & 7;            // K: rows 0..15
  int kr1 = 16 + (tid >> 3),   ks1 = tid & 7;            // K: rows 16..31
  int vr0 = tid >> 2,          vs0 = tid & 3;            // V: rows 0..31
  int vr1 = 32 + (tid >> 2),   vs1 = tid & 3;            // V: rows 32..63

  // prologue: stage tile 0 into buffer 0
  copy16(K + (size_t)kr0 * DATTN + ks0 * 8, &Kt_s[0][kr0 * 64 + ks0 * 8]);
  copy16(K + (size_t)kr1 * DATTN + ks1 * 8, &Kt_s[0][kr1 * 64 + ks1 * 8]);
  copy16(Vg + (size_t)vr0 * SEQ + vs0 * 8,  &Vt_s[0][vr0 * 32 + vs0 * 8]);
  copy16(Vg + (size_t)vr1 * SEQ + vs1 * 8,  &Vt_s[0][vr1 * 32 + vs1 * 8]);
  wait_async();
  __syncthreads();

  for (int t0 = 0; t0 < SEQ; t0 += 32) {
    int cur = (t0 >> 5) & 1;
    if (t0 + 32 < SEQ) {  // issue next tile into the other buffer (overlaps compute)
      int nx = cur ^ 1, t1 = t0 + 32;
      copy16(K + (size_t)(t1 + kr0) * DATTN + ks0 * 8, &Kt_s[nx][kr0 * 64 + ks0 * 8]);
      copy16(K + (size_t)(t1 + kr1) * DATTN + ks1 * 8, &Kt_s[nx][kr1 * 64 + ks1 * 8]);
      copy16(Vg + (size_t)vr0 * SEQ + t1 + vs0 * 8,    &Vt_s[nx][vr0 * 32 + vs0 * 8]);
      copy16(Vg + (size_t)vr1 * SEQ + t1 + vs1 * 8,    &Vt_s[nx][vr1 * 32 + vs1 * 8]);
    }
    const bf16* Kc = Kt_s[cur];
    const bf16* Vc = Vt_s[cur];

    // scores: two 16x16 f32 tiles (32 keys), dh=64 in two K-steps, B from LDS
    v8f s0 = {}, s1 = {};
    s0 = wmma_bf16(qa0, *(const v16bf*)&Kc[(n)      * 64 + 0  + 16 * half], s0);
    s0 = wmma_bf16(qa1, *(const v16bf*)&Kc[(n)      * 64 + 32 + 16 * half], s0);
    s1 = wmma_bf16(qa0, *(const v16bf*)&Kc[(16 + n) * 64 + 0  + 16 * half], s1);
    s1 = wmma_bf16(qa1, *(const v16bf*)&Kc[(16 + n) * 64 + 32 + 16 * half], s1);

#pragma unroll
    for (int r = 0; r < 8; ++r) {
      float a0 = s0[r] * sc, a1 = s1[r] * sc;
      float v = fmaxf(a0, a1);
      v = fmaxf(v, __shfl_xor(v, 1, 32));
      v = fmaxf(v, __shfl_xor(v, 2, 32));
      v = fmaxf(v, __shfl_xor(v, 4, 32));
      v = fmaxf(v, __shfl_xor(v, 8, 32));      // row max within 16-lane half
      float mn = fmaxf(m[r], v);
      float f  = __expf(m[r] - mn);
      m[r] = mn;
      float p0 = __expf(a0 - mn), p1 = __expf(a1 - mn);
      float rs = p0 + p1;
      rs += __shfl_xor(rs, 1, 32);
      rs += __shfl_xor(rs, 2, 32);
      rs += __shfl_xor(rs, 4, 32);
      rs += __shfl_xor(rs, 8, 32);
      l[r] = l[r] * f + rs;
      o0[r] *= f; o1[r] *= f; o2[r] *= f; o3[r] *= f;
      int row = 8 * half + r;
      P[row * 32 + n]      = (bf16)p0;
      P[row * 32 + 16 + n] = (bf16)p1;
    }
    // per-wave DS write->read; DS ops are in-order within a wave
    v16bf pa = load_a_bf16(P, 32, 0, 0, lane);
    o0 = wmma_bf16(pa, *(const v16bf*)&Vc[(0  + n) * 32 + 16 * half], o0);
    o1 = wmma_bf16(pa, *(const v16bf*)&Vc[(16 + n) * 32 + 16 * half], o1);
    o2 = wmma_bf16(pa, *(const v16bf*)&Vc[(32 + n) * 32 + 16 * half], o2);
    o3 = wmma_bf16(pa, *(const v16bf*)&Vc[(48 + n) * 32 + 16 * half], o3);

    wait_async();        // next tile fully in LDS
    __syncthreads();     // everyone done reading 'cur' before it is overwritten
  }

  // normalize and scatter into AO (B, S, NACT*64) bf16
#pragma unroll
  for (int r = 0; r < 8; ++r) {
    float inv = 1.0f / l[r];
    int row = mt * 16 + 8 * half + r;
    bf16* dp = AO + ((size_t)b * SEQ + row) * (NACT * DATTN) + k * DATTN + n;
    dp[0]  = (bf16)(o0[r] * inv);
    dp[16] = (bf16)(o1[r] * inv);
    dp[32] = (bf16)(o2[r] * inv);
    dp[48] = (bf16)(o3[r] * inv);
  }
}

// ---------------------------------------------------------------- output projection
// (B*S x 256) bf16 @ Wot([1024][256] bf16) + bo -> f32; wave computes 16x64
__global__ __launch_bounds__(128) void k_oproj(
    const bf16* __restrict__ AO, const bf16* __restrict__ Wot,
    const float* __restrict__ bo, float* __restrict__ out) {
  int wave = (blockIdx.x * blockDim.x + threadIdx.x) >> 5;
  int lane = threadIdx.x & 31;
  int ng = wave & 15;                // 16 groups of 64 cols across 1024
  int mt = wave >> 4;                // 512 tiles of 16 across B*S

  v8f acc[4] = {};
#pragma unroll
  for (int kk = 0; kk < NACT * DATTN; kk += 32) {
    v16bf a = load_a_bf16(AO, NACT * DATTN, mt * 16, kk, lane);
#pragma unroll
    for (int j = 0; j < 4; ++j) {
      v16bf bm = load_b_kcontig(Wot, NACT * DATTN, kk, ng * 64 + j * 16, lane);
      acc[j] = wmma_bf16(a, bm, acc[j]);
    }
  }
  int half = lane >> 4, n = lane & 15;
#pragma unroll
  for (int j = 0; j < 4; ++j) {
    int col = ng * 64 + j * 16 + n;
    float bb = bo[col];
#pragma unroll
    for (int r = 0; r < 8; ++r) {
      int row = mt * 16 + 8 * half + r;
      // streamed once, never re-read: non-temporal store
      __builtin_nontemporal_store(acc[j][r] + bb, &out[(size_t)row * DMODEL + col]);
    }
  }
}

// ---------------------------------------------------------------- launch
extern "C" void kernel_launch(void* const* d_in, const int* in_sizes, int n_in,
                              void* d_out, int out_size, void* d_ws, size_t ws_size,
                              hipStream_t stream) {
  const float* x  = (const float*)d_in[0];
  const float* Wq = (const float*)d_in[1];
  const float* bq = (const float*)d_in[2];
  const float* Wk = (const float*)d_in[3];
  const float* bk = (const float*)d_in[4];
  const float* Wv = (const float*)d_in[5];
  const float* bv = (const float*)d_in[6];
  const float* Wr = (const float*)d_in[7];
  const float* br = (const float*)d_in[8];
  const float* Wo = (const float*)d_in[9];
  const float* bo = (const float*)d_in[10];
  float* out = (float*)d_out;

  char* w = (char*)d_ws;
  const size_t XB_BYTES  = (size_t)BATCH * SEQ * DMODEL * 2;          // 16 MB
  const size_t QKV_BYTES = (size_t)BATCH * NACT * SEQ * DATTN * 2;    // 4 MB each
  const size_t WG_BYTES  = (size_t)BATCH * 3 * 256 * DMODEL * 2;      // 3 MB

  float* meanx  = (float*)(w);
  int*   topidx = (int*)  (w + 16384);
  float* gate   = (float*)(w + 16384 + 256);
  bf16*  xb     = (bf16*) (w + 32768);
  bf16*  Qs     = (bf16*) ((char*)xb + XB_BYTES);
  bf16*  Ks     = (bf16*) ((char*)Qs + QKV_BYTES);
  bf16*  Vt     = (bf16*) ((char*)Ks + QKV_BYTES);
  bf16*  AO     = (bf16*) ((char*)Vt + QKV_BYTES);
  bf16*  Wg     = (bf16*) ((char*)AO + QKV_BYTES);
  bf16*  Wot    = (bf16*) ((char*)Wg + WG_BYTES);
  (void)in_sizes; (void)n_in; (void)out_size; (void)ws_size;

  k_meanx  <<<(BATCH * DMODEL) / 256, 256, 0, stream>>>(x, meanx);
  k_router <<<1, 64, 0, stream>>>(meanx, Wr, br, topidx, gate);
  k_xcvt   <<<(BATCH * SEQ * DMODEL) / (256 * 8), 256, 0, stream>>>(x, xb);
  k_wgather<<<(BATCH * 3 * 256 * 256) / 256, 256, 0, stream>>>(Wq, Wk, Wv, topidx, Wg);
  k_wot    <<<(DMODEL * 64) / 256, 256, 0, stream>>>(Wo, Wot);
  // 24 combos * 256 mtiles = 6144 waves -> 1536 blocks of 4 waves
  k_qkv    <<<1536, 128, 0, stream>>>(xb, Wg, bq, bk, bv, topidx, gate, Qs, Ks, Vt);
  // 8 combos * 64 blocks (4 qtiles each) = 512 blocks
  k_attn   <<<512, 128, 0, stream>>>(Qs, Ks, Vt, AO);
  // 512 mtiles * 16 ngroups = 8192 waves -> 2048 blocks
  k_oproj  <<<2048, 128, 0, stream>>>(AO, Wot, bo, out);
}
